// SLSTM_68908455297204
// MI455X (gfx1250) — matrix operand order repaired and checked by reading, not compiled, not run on hardware
//
#include <hip/hip_runtime.h>
#include <hip/hip_bf16.h>

// ---------------------------------------------------------------------------
// sLSTM stack for MI455X (gfx1250), wave32 + WMMA bf16.
//   D=1024, H=1024 (4 heads x 256), 4H=4096, P=1365, L=1024, B=8, NL=4.
// Weights pre-packed once per call into bf16 WMMA B-fragments (L2-resident).
// 3 kernels per (layer, timestep): gates(+LN+state+GN), up(+GELU), down(+res).
// ---------------------------------------------------------------------------

typedef __bf16 bf16_t;
typedef __attribute__((ext_vector_type(16))) __bf16 v16bf;
typedef __attribute__((ext_vector_type(8)))  float  v8f;

#define D_    1024
#define H_    1024
#define HS_   256
#define NH_   4
#define NL_   4
#define B_    8
#define L_    1024
#define P_    1365
#define TWOP_ 2730
#define PT_   86        // ceil(P/16)
#define KDN_  43        // ceil(P/32)  (K tiles for down GEMM, padded to 1376)
#define EPSF  1e-5f

__device__ __forceinline__ v8f wmma_bf16(v16bf a, v16bf b, v8f c) {
  return __builtin_amdgcn_wmma_f32_16x16x32_bf16(false, a, false, b, (short)0, c,
                                                 false, false);
}

// A-fragment (16x32, bf16) intra-tile K offset for (lane, j): k = 16*(lane/16) + j
__device__ __forceinline__ int koffA(int lane, int j) {
  return j + ((lane >> 4) << 4);
}

// ---------------------------------------------------------------------------
// Weight packing: B-fragment tiles (32x16). Lane l<16: N=l, K=j; l>=16: K=16+j.
// dst layout: [nt][kt][lane*16+j]
// ---------------------------------------------------------------------------
__global__ void pack_b_kernel(const float* __restrict__ src, bf16_t* __restrict__ dst,
                              int ktiles, int ntiles, int Kvalid, int Nvalid, int ld) {
  int total = ktiles * ntiles * 512;
  int per = ktiles * 512;
  for (int idx = blockIdx.x * blockDim.x + threadIdx.x; idx < total;
       idx += gridDim.x * blockDim.x) {
    int nt = idx / per;
    int rem = idx - nt * per;
    int kt = rem >> 9;
    int lane = (rem >> 4) & 31;
    int j = rem & 15;
    int k = kt * 32 + j + ((lane >> 4) << 4);
    int n = nt * 16 + (lane & 15);
    float v = (k < Kvalid && n < Nvalid) ? src[(size_t)k * ld + n] : 0.0f;
    dst[idx] = (bf16_t)v;
  }
}

// up_w (H x 2P) packed as [pt][half][kt][512]; half0 = u1 cols, half1 = u2 cols.
__global__ void pack_up_kernel(const float* __restrict__ src, bf16_t* __restrict__ dst) {
  const int total = PT_ * 2 * 32 * 512;
  for (int idx = blockIdx.x * blockDim.x + threadIdx.x; idx < total;
       idx += gridDim.x * blockDim.x) {
    int pt = idx >> 15;                 // / (2*32*512)
    int rem = idx & 32767;
    int half = rem >> 14;
    int rem2 = rem & 16383;
    int kt = rem2 >> 9;
    int lane = (rem2 >> 4) & 31;
    int j = rem2 & 15;
    int k = kt * 32 + j + ((lane >> 4) << 4);     // k < 1024 always
    int col = pt * 16 + (lane & 15);
    float v = (col < P_) ? src[(size_t)k * TWOP_ + half * P_ + col] : 0.0f;
    dst[idx] = (bf16_t)v;
  }
}

// ---------------------------------------------------------------------------
// Kernel 1: LN(x_t) -> xn ; gx = xn@W_i, gh = h@W_h (this head) via WMMA ;
// fused sLSTM state update ; per-head groupnorm ; pack y into bf16 A-frags.
// grid = 4 blocks (one per head), 512 threads (16 waves; wave = hs-tile).
// ---------------------------------------------------------------------------
__global__ __launch_bounds__(512) void slstm_gates_kernel(
    const float* __restrict__ x_t, long x_bstride,
    const float* __restrict__ ln_w, const float* __restrict__ ln_b,
    const bf16_t* __restrict__ wi_pack,   // layer base: [nt(256)][kt(32)][512]
    const bf16_t* __restrict__ wh_pack,   // layer base: [head][nt(64)][kt(8)][512]
    const float* __restrict__ b_i, const float* __restrict__ b_h,
    const float* __restrict__ gn_w, const float* __restrict__ gn_b,
    float* __restrict__ h_state, float* __restrict__ c_state,
    float* __restrict__ m_state, float* __restrict__ n_state,
    bf16_t* __restrict__ yfrag) {
  __shared__ __align__(32) bf16_t xnf[32 * 512];
  __shared__ __align__(32) bf16_t hf[8 * 512];
  __shared__ float red[16][8][2];
  __shared__ float s_mu[8], s_rs[8], s_gmu[8], s_grs[8];

  const int tid = threadIdx.x;
  const int wave = tid >> 5;
  const int lane = tid & 31;
  const int head = blockIdx.x;

  // --- Phase 1: LayerNorm stats, one wave per batch row -------------------
  if (wave < 8) {
    const float* xr = x_t + (size_t)wave * x_bstride;
    float s = 0.f, s2 = 0.f;
#pragma unroll
    for (int i = 0; i < 32; ++i) {
      float v = xr[lane + i * 32];
      s += v;
      s2 += v * v;
    }
#pragma unroll
    for (int off = 16; off > 0; off >>= 1) {
      s += __shfl_xor(s, off, 32);
      s2 += __shfl_xor(s2, off, 32);
    }
    if (lane == 0) {
      float mu = s * (1.f / 1024.f);
      float var = s2 * (1.f / 1024.f) - mu * mu;
      s_mu[wave] = mu;
      s_rs[wave] = rsqrtf(var + EPSF);
    }
  }
  __syncthreads();

  // --- Phase 2: build bf16 A-fragments in LDS (xn: 32 tiles, h: 8 tiles) --
  for (int idx = tid; idx < 32 * 512; idx += 512) {
    int kt = idx >> 9;
    int la = (idx >> 4) & 31;
    int j = idx & 15;
    int m = la & 15;
    int k = kt * 32 + koffA(la, j);
    float v = 0.f;
    if (m < B_)
      v = (x_t[(size_t)m * x_bstride + k] - s_mu[m]) * s_rs[m] * ln_w[k] + ln_b[k];
    xnf[idx] = (bf16_t)v;
  }
  for (int idx = tid; idx < 8 * 512; idx += 512) {
    int kt = idx >> 9;
    int la = (idx >> 4) & 31;
    int j = idx & 15;
    int m = la & 15;
    int k = kt * 32 + koffA(la, j);
    float v = (m < B_) ? h_state[m * H_ + head * HS_ + k] : 0.f;
    hf[idx] = (bf16_t)v;
  }
  __syncthreads();

  // --- Phase 3: WMMA. Each wave: 1 hs-tile (16 cols) x 4 gates ------------
  const int ht = wave;  // 0..15
  v8f acc[4];
#pragma unroll
  for (int g = 0; g < 4; ++g) {
#pragma unroll
    for (int r = 0; r < 8; ++r) acc[g][r] = 0.f;
  }

#pragma unroll 4
  for (int kt = 0; kt < 32; ++kt) {
    v16bf a = *(const v16bf*)(xnf + kt * 512 + lane * 16);
    if (kt + 1 < 32)
      __builtin_prefetch(wi_pack + ((size_t)((head * 16 + ht) * 32 + kt + 1) << 9) +
                             lane * 16, 0, 3);
#pragma unroll
    for (int g = 0; g < 4; ++g) {
      const bf16_t* bp =
          wi_pack + ((size_t)(((g * 64 + head * 16 + ht) * 32) + kt) << 9) + lane * 16;
      v16bf b = *(const v16bf*)bp;
      acc[g] = wmma_bf16(a, b, acc[g]);
    }
  }
#pragma unroll
  for (int kt = 0; kt < 8; ++kt) {
    v16bf a = *(const v16bf*)(hf + kt * 512 + lane * 16);
#pragma unroll
    for (int g = 0; g < 4; ++g) {
      const bf16_t* bp =
          wh_pack + ((size_t)(((head * 64 + g * 16 + ht) * 8) + kt) << 9) + lane * 16;
      v16bf b = *(const v16bf*)bp;
      acc[g] = wmma_bf16(a, b, acc[g]);
    }
  }

  // --- Phase 4: fused sLSTM state update (lanes 0..15 hold rows m=0..7) ---
  const int ncol = lane & 15;
  const int hsc = ht * 16 + ncol;       // 0..255 within head
  const int colH = head * HS_ + hsc;    // global H column
  float bias[4];
#pragma unroll
  for (int g = 0; g < 4; ++g) {
    int c = g * H_ + colH;
    bias[g] = b_i[c] + b_h[c];
  }
  float hv[8], srow[8], s2row[8];
#pragma unroll
  for (int r = 0; r < 8; ++r) { hv[r] = 0.f; srow[r] = 0.f; s2row[r] = 0.f; }

  if (lane < 16) {
#pragma unroll
    for (int r = 0; r < 8; ++r) {
      int sidx = r * H_ + colH;
      float it = acc[0][r] + bias[0];
      float ft = acc[1][r] + bias[1];
      float zt = acc[2][r] + bias[2];
      float ot = acc[3][r] + bias[3];
      float mp = m_state[sidx];
      float cp = c_state[sidx];
      float np = n_state[sidx];
      float mn = fmaxf(ft + mp, it);
      float iv = expf(it - mn);
      float fv = expf(ft - mn + mp);
      float zv = tanhf(zt);
      float ov = 1.f / (1.f + expf(-ot));
      float cn = fv * cp + iv * zv;
      float nn = fv * np + iv;
      float hn = ov * (cn / nn);
      m_state[sidx] = mn;
      c_state[sidx] = cn;
      n_state[sidx] = nn;
      h_state[sidx] = hn;
      hv[r] = hn;
      srow[r] = hn;
      s2row[r] = hn * hn;
    }
  }
  // reduce over this wave's 16 columns (within lanes 0..15)
#pragma unroll
  for (int r = 0; r < 8; ++r) {
#pragma unroll
    for (int off = 8; off > 0; off >>= 1) {
      srow[r] += __shfl_xor(srow[r], off, 16);
      s2row[r] += __shfl_xor(s2row[r], off, 16);
    }
  }
  if (lane == 0) {
#pragma unroll
    for (int r = 0; r < 8; ++r) {
      red[wave][r][0] = srow[r];
      red[wave][r][1] = s2row[r];
    }
  }
  __syncthreads();
  if (tid < 8) {
    float s = 0.f, s2 = 0.f;
#pragma unroll
    for (int w = 0; w < 16; ++w) {
      s += red[w][tid][0];
      s2 += red[w][tid][1];
    }
    float mu = s * (1.f / 256.f);
    float var = s2 * (1.f / 256.f) - mu * mu;
    s_gmu[tid] = mu;
    s_grs[tid] = rsqrtf(var + EPSF);
  }
  __syncthreads();

  // --- Phase 5: y = GN(h)*gn_w+gn_b, scatter into bf16 A-fragments --------
  {
    float gw = gn_w[colH], gb = gn_b[colH];
    int kt = colH >> 5;
    int kk = colH & 31;
    int lhalf = kk >> 4;
    int j = kk & 15;
#pragma unroll
    for (int r = 0; r < 8; ++r) {
      int m = r + ((lane >> 4) << 3);  // lanes>=16 fill zero pad rows 8..15
      float y = 0.f;
      if (lane < 16) y = (hv[r] - s_gmu[r]) * s_grs[r] * gw + gb;
      yfrag[(size_t)kt * 512 + (size_t)(m + (lhalf << 4)) * 16 + j] = (bf16_t)y;
    }
  }
}

// ---------------------------------------------------------------------------
// Kernel 2: up GEMM (u1,u2 tiles per wave) + u1*gelu(u2), pack act A-frags.
// grid 11 x 256 (88 waves, 86 active p-tiles).
// ---------------------------------------------------------------------------
__global__ __launch_bounds__(256) void slstm_up_kernel(
    const bf16_t* __restrict__ yfrag,
    const bf16_t* __restrict__ up_pack,  // layer base: [pt][2][kt(32)][512]
    const float* __restrict__ up_b,      // layer base (2P)
    bf16_t* __restrict__ actfrag) {
  const int tid = threadIdx.x;
  const int lane = tid & 31;
  const int wid = blockIdx.x * 8 + (tid >> 5);
  if (wid >= PT_) return;
  const int pt = wid;

  v8f acc1, acc2;
#pragma unroll
  for (int r = 0; r < 8; ++r) { acc1[r] = 0.f; acc2[r] = 0.f; }

#pragma unroll 4
  for (int kt = 0; kt < 32; ++kt) {
    v16bf a = *(const v16bf*)(yfrag + kt * 512 + lane * 16);
    v16bf b1 = *(const v16bf*)(up_pack + ((size_t)((pt * 2 + 0) * 32 + kt) << 9) + lane * 16);
    v16bf b2 = *(const v16bf*)(up_pack + ((size_t)((pt * 2 + 1) * 32 + kt) << 9) + lane * 16);
    acc1 = wmma_bf16(a, b1, acc1);
    acc2 = wmma_bf16(a, b2, acc2);
  }

  const int ncol = lane & 15;
  const int col = pt * 16 + ncol;  // 0..1375
  const bool valid = (col < P_);
  float bu1 = 0.f, bu2 = 0.f;
  if (valid) {
    bu1 = up_b[col];
    bu2 = up_b[P_ + col];
  }
  int kt = col >> 5;
  int kk = col & 31;
  int lhalf = kk >> 4;
  int j = kk & 15;
#pragma unroll
  for (int r = 0; r < 8; ++r) {
    int m = r + ((lane >> 4) << 3);
    float v = 0.f;
    if (lane < 16 && valid) {
      float u1 = acc1[r] + bu1;
      float u2 = acc2[r] + bu2;
      float g = 0.5f * u2 * (1.f + erff(u2 * 0.70710678118654752f));  // exact GELU
      v = u1 * g;
    }
    actfrag[(size_t)kt * 512 + (size_t)(m + (lhalf << 4)) * 16 + j] = (bf16_t)v;
  }
}

// ---------------------------------------------------------------------------
// Kernel 3: down GEMM + down_b + residual. grid 8 x 256 (64 waves = 64 n-tiles).
// ---------------------------------------------------------------------------
__global__ __launch_bounds__(256) void slstm_down_kernel(
    const bf16_t* __restrict__ actfrag,
    const bf16_t* __restrict__ dn_pack,  // layer base: [nt(64)][kt(43)][512]
    const float* __restrict__ down_b,    // layer base (D)
    const float* __restrict__ x_t, long x_bstride,
    float* __restrict__ out_t, long out_bstride) {
  const int tid = threadIdx.x;
  const int lane = tid & 31;
  const int nt = blockIdx.x * 8 + (tid >> 5);  // 0..63

  v8f acc;
#pragma unroll
  for (int r = 0; r < 8; ++r) acc[r] = 0.f;

  for (int kt = 0; kt < KDN_; ++kt) {
    v16bf a = *(const v16bf*)(actfrag + kt * 512 + lane * 16);
    v16bf b = *(const v16bf*)(dn_pack + ((size_t)(nt * KDN_ + kt) << 9) + lane * 16);
    acc = wmma_bf16(a, b, acc);
  }

  if (lane < 16) {
    int col = nt * 16 + lane;
    float db = down_b[col];
#pragma unroll
    for (int r = 0; r < 8; ++r) {
      float v = acc[r] + db + x_t[(size_t)r * x_bstride + col];
      out_t[(size_t)r * out_bstride + col] = v;
    }
  }
}

// 4-way state copy (init from h0/c0/m0/n0 and final save into d_out)
__global__ void copy4_kernel(const float* __restrict__ a, const float* __restrict__ b,
                             const float* __restrict__ c, const float* __restrict__ d,
                             float* __restrict__ oa, float* __restrict__ ob,
                             float* __restrict__ oc, float* __restrict__ od, int n) {
  int i = blockIdx.x * blockDim.x + threadIdx.x;
  if (i < n) {
    oa[i] = a[i];
    ob[i] = b[i];
    oc[i] = c[i];
    od[i] = d[i];
  }
}

// ---------------------------------------------------------------------------
extern "C" void kernel_launch(void* const* d_in, const int* in_sizes, int n_in,
                              void* d_out, int out_size, void* d_ws, size_t ws_size,
                              hipStream_t stream) {
  (void)in_sizes; (void)n_in; (void)out_size; (void)ws_size;
  const float* X   = (const float*)d_in[0];
  const float* H0  = (const float*)d_in[1];
  const float* C0  = (const float*)d_in[2];
  const float* M0  = (const float*)d_in[3];
  const float* N0  = (const float*)d_in[4];
  const float* LNW = (const float*)d_in[5];
  const float* LNB = (const float*)d_in[6];
  const float* WI  = (const float*)d_in[7];
  const float* BI  = (const float*)d_in[8];
  const float* WH  = (const float*)d_in[9];
  const float* BH  = (const float*)d_in[10];
  const float* GNW = (const float*)d_in[11];
  const float* GNB = (const float*)d_in[12];
  const float* UPW = (const float*)d_in[13];
  const float* UPB = (const float*)d_in[14];
  const float* DNW = (const float*)d_in[15];
  const float* DNB = (const float*)d_in[16];
  float* OUT = (float*)d_out;

  // per-layer packed-weight element counts
  const size_t WI_L = 256ull * 32 * 512;       // 4,194,304
  const size_t WH_L = 4ull * 64 * 8 * 512;     // 1,048,576
  const size_t UP_L = (size_t)PT_ * 2 * 32 * 512;
  const size_t DN_L = 64ull * KDN_ * 512;

  char* ws = (char*)d_ws;
  size_t off = 0;
  auto carve = [&](size_t bytes) {
    size_t o = off;
    off = (off + bytes + 255) & ~(size_t)255;
    return o;
  };
  bf16_t* wi_pack = (bf16_t*)(ws + carve(NL_ * WI_L * sizeof(bf16_t)));
  bf16_t* wh_pack = (bf16_t*)(ws + carve(NL_ * WH_L * sizeof(bf16_t)));
  bf16_t* up_pack = (bf16_t*)(ws + carve(NL_ * UP_L * sizeof(bf16_t)));
  bf16_t* dn_pack = (bf16_t*)(ws + carve(NL_ * DN_L * sizeof(bf16_t)));
  bf16_t* yfrag   = (bf16_t*)(ws + carve(32ull * 512 * sizeof(bf16_t)));
  bf16_t* actfrag = (bf16_t*)(ws + carve((size_t)KDN_ * 512 * sizeof(bf16_t)));
  float* h_st = (float*)(ws + carve((size_t)B_ * H_ * sizeof(float)));
  float* c_st = (float*)(ws + carve((size_t)B_ * H_ * sizeof(float)));
  float* m_st = (float*)(ws + carve((size_t)B_ * H_ * sizeof(float)));
  float* n_st = (float*)(ws + carve((size_t)B_ * H_ * sizeof(float)));
  float* seqA = (float*)(ws + carve((size_t)L_ * B_ * D_ * sizeof(float)));
  float* seqB = (float*)(ws + carve((size_t)L_ * B_ * D_ * sizeof(float)));

  // ---- one-time (per call) weight packing into bf16 WMMA fragments -------
  for (int l = 0; l < NL_; ++l) {
    pack_b_kernel<<<2048, 256, 0, stream>>>(WI + (size_t)l * D_ * 4096,
                                            wi_pack + (size_t)l * WI_L,
                                            32, 256, 1024, 4096, 4096);
    for (int hd = 0; hd < NH_; ++hd)
      pack_b_kernel<<<512, 256, 0, stream>>>(
          WH + (size_t)(l * NH_ + hd) * HS_ * 1024,
          wh_pack + (size_t)(l * NH_ + hd) * 64 * 8 * 512,
          8, 64, 256, 1024, 1024);
    pack_up_kernel<<<2048, 256, 0, stream>>>(UPW + (size_t)l * H_ * TWOP_,
                                             up_pack + (size_t)l * UP_L);
    pack_b_kernel<<<1024, 256, 0, stream>>>(DNW + (size_t)l * P_ * D_,
                                            dn_pack + (size_t)l * DN_L,
                                            KDN_, 64, P_, 1024, 1024);
  }

  const size_t state_out = (size_t)B_ * L_ * D_;  // start of h-stack in d_out
  for (int l = 0; l < NL_; ++l) {
    // init running state from h0/c0/m0/n0 layer slices
    copy4_kernel<<<32, 256, 0, stream>>>(H0 + (size_t)l * B_ * H_,
                                         C0 + (size_t)l * B_ * H_,
                                         M0 + (size_t)l * B_ * H_,
                                         N0 + (size_t)l * B_ * H_,
                                         h_st, c_st, m_st, n_st, B_ * H_);
    const float* lnw = LNW + (size_t)l * D_;
    const float* lnb = LNB + (size_t)l * D_;
    const float* bi  = BI + (size_t)l * 4096;
    const float* bh  = BH + (size_t)l * 4096;
    const float* gnw = GNW + (size_t)l * H_;
    const float* gnb = GNB + (size_t)l * H_;
    const float* upb = UPB + (size_t)l * TWOP_;
    const float* dnb = DNB + (size_t)l * D_;
    const bf16_t* wip = wi_pack + (size_t)l * WI_L;
    const bf16_t* whp = wh_pack + (size_t)l * WH_L;
    const bf16_t* upp = up_pack + (size_t)l * UP_L;
    const bf16_t* dnp = dn_pack + (size_t)l * DN_L;

    // ping-pong sequence buffers: l0: X->A, l1: A->B, l2: B->A, l3: A->OUT
    const float* inb = (l & 1) ? seqA : seqB;  // only used for l >= 1
    float* outb = (l & 1) ? seqB : seqA;       // only used for l < 3

    for (int t = 0; t < L_; ++t) {
      const float* x_t;
      long xbs;
      if (l == 0) {
        x_t = X + (size_t)t * D_;     // x[b][t][d], b-stride = L*D
        xbs = (long)L_ * D_;
      } else {
        x_t = inb + (size_t)t * B_ * D_;
        xbs = D_;
      }
      slstm_gates_kernel<<<NH_, 512, 0, stream>>>(
          x_t, xbs, lnw, lnb, wip, whp, bi, bh, gnw, gnb,
          h_st, c_st, m_st, n_st, yfrag);
      slstm_up_kernel<<<11, 256, 0, stream>>>(yfrag, upp, upb, actfrag);
      float* o_t;
      long obs;
      if (l == NL_ - 1) {
        o_t = OUT + (size_t)t * D_;   // out[b][t][d]
        obs = (long)L_ * D_;
      } else {
        o_t = outb + (size_t)t * B_ * D_;
        obs = D_;
      }
      slstm_down_kernel<<<8, 256, 0, stream>>>(actfrag, dnp, dnb, x_t, xbs, o_t, obs);
    }
    // save final (h, c, m, n) for this layer into d_out tail
    copy4_kernel<<<32, 256, 0, stream>>>(
        h_st, c_st, m_st, n_st,
        OUT + state_out + (size_t)l * B_ * H_,
        OUT + state_out + (size_t)NL_ * B_ * H_ + (size_t)l * B_ * H_,
        OUT + state_out + 2ull * NL_ * B_ * H_ + (size_t)l * B_ * H_,
        OUT + state_out + 3ull * NL_ * B_ * H_ + (size_t)l * B_ * H_, B_ * H_);
  }
}
